// Attention_6932077216322
// MI455X (gfx1250) — matrix operand order repaired and checked by reading, not compiled, not run on hardware
//
#include <hip/hip_runtime.h>

typedef __attribute__((ext_vector_type(16))) __bf16 v16bf;
typedef __attribute__((ext_vector_type(8)))  float  v8f;

#define NB 4
#define SS 2048
#define HH 1024
#define NHD 4
#define DD 256
#define MASK_FILL_F (-4294967295.0f)
#define LN_EPS_F 1e-5f

// async global->LDS 16B per lane (CDNA5): dst = per-lane LDS byte offset, src = per-lane global addr
#define ASYNC_LD16(ldsoff, gptr)                                                   \
    asm volatile("global_load_async_to_lds_b128 %0, %1, off"                       \
                 :: "v"(ldsoff), "v"(gptr) : "memory")
#define WAIT_ASYNC0() asm volatile("s_wait_asynccnt 0x0" ::: "memory")

static __device__ __forceinline__ unsigned lds_off(const void* p) {
    return (unsigned)(unsigned long long)p;  // generic shared addr: low 32 bits = LDS offset
}

// ---------------------------------------------------------------- converts

__global__ void cvt_bf16_kernel(const float* __restrict__ x, __bf16* __restrict__ y, int n4) {
    int i = blockIdx.x * 256 + threadIdx.x;
    if (i >= n4) return;
    int j = i * 4;
    float4 f = *(const float4*)(x + j);
    y[j + 0] = (__bf16)f.x; y[j + 1] = (__bf16)f.y;
    y[j + 2] = (__bf16)f.z; y[j + 3] = (__bf16)f.w;
}

// decoder_input f32 [M][1024] -> bf16 into concat buffer [M][2048] cols 0..1023
__global__ void cvt_dec_kernel(const float* __restrict__ x, __bf16* __restrict__ y, int n4) {
    int i = blockIdx.x * 256 + threadIdx.x;
    if (i >= n4) return;
    int j = i * 4;
    int row = j >> 10, col = j & 1023;
    float4 f = *(const float4*)(x + j);
    __bf16* p = y + (size_t)row * 2048 + col;
    p[0] = (__bf16)f.x; p[1] = (__bf16)f.y; p[2] = (__bf16)f.z; p[3] = (__bf16)f.w;
}

// W f32 [N][K] -> WT bf16 [K][N]
__global__ void transpose_w_kernel(const float* __restrict__ W, __bf16* __restrict__ WT,
                                   int N, int K) {
    size_t i = (size_t)blockIdx.x * 256 + threadIdx.x;
    if (i >= (size_t)N * K) return;
    int n = (int)(i / K), k = (int)(i % K);
    WT[(size_t)k * N + n] = (__bf16)W[i];
}

// ---------------------------------------------------------------- WMMA GEMM
// C[M][N] = A[M][K] * B[K][N], bf16 row-major, f32 accumulate.
// Block tile 64x256, BK=32. 8 wave32s in 2(M)x4(N); each wave 32x64 = 2x4 accumulators.
// Double-buffered LDS staged with GLOBAL_LOAD_ASYNC_TO_LDS_B128 (ASYNCcnt): DMA of tile
// k+1 overlaps the WMMA stream on tile k; consumed via ds_load_b128 -> v_wmma.

#define BM 64
#define BN 256
#define BK 32
#define LDA_S 40    // padded LDS row stride (bf16), 80B: 16B-aligned, bank-spread
#define LDB_S 264   // 528B

struct GemmP {
    const __bf16* A;  const __bf16* Bm;
    size_t sAz, sBz;           // z-batch strides (elements)
    int lda, ldb, K;
    float* outF; __bf16* outBf;
    const unsigned char* mask; // bool tensor [B][S][S]
    const float* bias; const float* resid;
};

enum { EPI_HEAD = 0, EPI_HEAD_T = 1, EPI_SCORES = 2, EPI_CTX = 3, EPI_FINAL = 4 };

template <int EPI>
__global__ __launch_bounds__(256, 2) void gemm_bf16_kernel(GemmP p) {
    __shared__ __bf16 smA[2][BM * LDA_S];
    __shared__ __bf16 smB[2][BK * LDB_S];

    const int t = threadIdx.x;
    const int lane = t & 31;
    const int wave = t >> 5;
    const int wm = wave & 1;    // M strip (x32)
    const int wn = wave >> 1;   // N strip (x64)
    const int z = blockIdx.z;

    const __bf16* Ag = p.A + (size_t)z * p.sAz + (size_t)(blockIdx.y * BM) * p.lda;
    const __bf16* Bg = p.Bm + (size_t)z * p.sBz + (size_t)blockIdx.x * BN;

    // staging addresses (fixed per thread)
    const int ar = t >> 2, ac = (t & 3) * 8;              // A: 64 rows x 4 chunks
    const int br = t >> 3, bc = (t & 7) * 32;             // B: 32 rows x 8 thread-cols x 32 elems
    unsigned ldsAo[2], ldsBo[2];
#pragma unroll
    for (int u = 0; u < 2; ++u) {
        ldsAo[u] = lds_off(&smA[u][ar * LDA_S + ac]);
        ldsBo[u] = lds_off(&smB[u][br * LDB_S + bc]);
    }
    const __bf16* gA = Ag + (size_t)ar * p.lda + ac;
    const __bf16* gB = Bg + (size_t)br * p.ldb + bc;

    auto stage = [&](int buf, int kk) {
        ASYNC_LD16(ldsAo[buf], gA + kk);
#pragma unroll
        for (int c = 0; c < 4; ++c)
            ASYNC_LD16(ldsBo[buf] + c * 16u, gB + (size_t)kk * p.ldb + c * 8);
    };

    v8f acc[2][4];
#pragma unroll
    for (int i = 0; i < 2; ++i)
#pragma unroll
        for (int j = 0; j < 4; ++j) acc[i][j] = v8f{};

    // prologue: fill buffer 0
    stage(0, 0);
    WAIT_ASYNC0();
    __syncthreads();

    int cur = 0;
    for (int k0 = 0; k0 < p.K; k0 += BK) {
        if (k0 + BK < p.K) stage(cur ^ 1, k0 + BK);  // DMA next tile during compute

        // A operands (16x32 each): lane m<16 holds K{0..7,16..23}; lane m+16 holds K{8..15,24..31}
        v16bf a[2];
        {
            const int m = lane & 15;
            const int kb = (lane < 16) ? 0 : 8;
#pragma unroll
            for (int i = 0; i < 2; ++i) {
                const __bf16* pa = &smA[cur][(wm * 32 + i * 16 + m) * LDA_S];
                *(float4*)&a[i]       = *(const float4*)(pa + kb);
                *((float4*)&a[i] + 1) = *(const float4*)(pa + kb + 16);
            }
        }
#pragma unroll
        for (int j = 0; j < 4; ++j) {
            // B operand (32x16): lane l holds column N=l%16; lane<16 K{0..7,16..23}, lane>=16 K{8..15,24..31}
            v16bf b;
            const int kbb = (lane < 16) ? 0 : 8;
            const __bf16* pb = &smB[cur][wn * 64 + j * 16 + (lane & 15)];
#pragma unroll
            for (int s = 0; s < 8; ++s) {
                b[s]     = pb[(kbb + s) * LDB_S];
                b[s + 8] = pb[(kbb + 16 + s) * LDB_S];
            }
            acc[0][j] = __builtin_amdgcn_wmma_f32_16x16x32_bf16(
                false, a[0], false, b, (short)0, acc[0][j], false, false);
            acc[1][j] = __builtin_amdgcn_wmma_f32_16x16x32_bf16(
                false, a[1], false, b, (short)0, acc[1][j], false, false);
        }

        WAIT_ASYNC0();     // next tile's DMA complete (no-op on last iteration)
        __syncthreads();   // all waves done reading `cur` and all DMAs visible
        cur ^= 1;
    }

    // Epilogue. C layout: VGPR v, lanes 0-15 -> (M=v, N=lane); lanes 16-31 -> (M=v+8, N=lane-16)
#pragma unroll
    for (int i = 0; i < 2; ++i) {
#pragma unroll
        for (int j = 0; j < 4; ++j) {
            int n = blockIdx.x * BN + wn * 64 + j * 16 + (lane & 15);
#pragma unroll
            for (int v = 0; v < 8; ++v) {
                int m = blockIdx.y * BM + wm * 32 + i * 16 + v + ((lane >> 4) << 3);
                float c = acc[i][j][v];
                if constexpr (EPI == EPI_HEAD) {           // -> [h][b][s][d] bf16
                    int h = n >> 8, d = n & 255, b = m >> 11, s = m & 2047;
                    p.outBf[((((size_t)h * NB + b) * SS + s) << 8) + d] = (__bf16)c;
                } else if constexpr (EPI == EPI_HEAD_T) {  // -> [h][b][d][s] bf16 (K transposed)
                    int h = n >> 8, d = n & 255, b = m >> 11, s = m & 2047;
                    p.outBf[((((size_t)h * NB + b) * DD + d) << 11) + s] = (__bf16)c;
                } else if constexpr (EPI == EPI_SCORES) {  // scale + mask -> f32 attns
                    int b = z & 3;
                    c *= 0.0625f;  // 1/sqrt(256)
                    size_t mi = (((size_t)b * SS + m) << 11) + n;
                    if (p.mask[mi]) c = MASK_FILL_F;
                    p.outF[(((size_t)z * SS + m) << 11) + n] = c;
                } else if constexpr (EPI == EPI_CTX) {     // -> concat[:,1024+h*256+d] bf16
                    int b = z & 3, h = z >> 2;
                    p.outBf[(((size_t)b * SS + m) << 11) + 1024 + h * DD + n] = (__bf16)c;
                } else {                                   // EPI_FINAL: +bias +residual -> f32
                    size_t o = ((size_t)m << 10) + n;
                    p.outF[o] = c + p.bias[n] + p.resid[o];
                }
            }
        }
    }
}

// ---------------------------------------------------------------- softmax rows
__global__ __launch_bounds__(256) void softmax_kernel(float* __restrict__ attnF,
                                                      __bf16* __restrict__ Pb,
                                                      const float* __restrict__ qmask) {
    __shared__ float red[256];
    int row = blockIdx.x;
    int q = row & (SS - 1);
    int b = (row >> 11) & 3;
    float* p = attnF + ((size_t)row << 11);
    int t = threadIdx.x;

    float v[8], mx = -3.4e38f;
#pragma unroll
    for (int i = 0; i < 8; ++i) { v[i] = p[t + 256 * i]; mx = fmaxf(mx, v[i]); }
    red[t] = mx; __syncthreads();
    for (int s = 128; s > 0; s >>= 1) { if (t < s) red[t] = fmaxf(red[t], red[t + s]); __syncthreads(); }
    mx = red[0]; __syncthreads();

    float sum = 0.f;
#pragma unroll
    for (int i = 0; i < 8; ++i) { v[i] = __expf(v[i] - mx); sum += v[i]; }
    red[t] = sum; __syncthreads();
    for (int s = 128; s > 0; s >>= 1) { if (t < s) red[t] += red[t + s]; __syncthreads(); }

    float scale = qmask[b * SS + q] / red[0];
#pragma unroll
    for (int i = 0; i < 8; ++i) {
        float o = v[i] * scale;
        p[t + 256 * i] = o;
        Pb[((size_t)row << 11) + t + 256 * i] = (__bf16)o;
    }
}

// ---------------------------------------------------------------- layernorm (in place)
__global__ __launch_bounds__(256) void layernorm_kernel(float* __restrict__ out,
                                                        const float* __restrict__ scale,
                                                        const float* __restrict__ bias) {
    __shared__ float red[256];
    int row = blockIdx.x, t = threadIdx.x;
    float* p = out + ((size_t)row << 10);
    float v[4], s = 0.f;
#pragma unroll
    for (int i = 0; i < 4; ++i) { v[i] = p[t + 256 * i]; s += v[i]; }
    red[t] = s; __syncthreads();
    for (int k = 128; k > 0; k >>= 1) { if (t < k) red[t] += red[t + k]; __syncthreads(); }
    float mu = red[0] * (1.0f / HH); __syncthreads();
    s = 0.f;
#pragma unroll
    for (int i = 0; i < 4; ++i) { float d = v[i] - mu; s += d * d; }
    red[t] = s; __syncthreads();
    for (int k = 128; k > 0; k >>= 1) { if (t < k) red[t] += red[t + k]; __syncthreads(); }
    float rstd = rsqrtf(red[0] * (1.0f / HH) + LN_EPS_F);
#pragma unroll
    for (int i = 0; i < 4; ++i) {
        int n = t + 256 * i;
        p[n] = (v[i] - mu) * rstd * scale[n] + bias[n];
    }
}

// ---------------------------------------------------------------- launch

extern "C" void kernel_launch(void* const* d_in, const int* in_sizes, int n_in,
                              void* d_out, int out_size, void* d_ws, size_t ws_size,
                              hipStream_t stream) {
    const float* memory  = (const float*)d_in[0];
    const float* decoder = (const float*)d_in[1];
    const unsigned char* mask = (const unsigned char*)d_in[2];  // bool
    const float* qmask   = (const float*)d_in[3];
    const float* Wk = (const float*)d_in[4];
    const float* Wv = (const float*)d_in[5];
    const float* Wq = (const float*)d_in[6];
    const float* Wf = (const float*)d_in[7];
    const float* bf = (const float*)d_in[8];
    const float* lns = (const float*)d_in[9];
    const float* lnb = (const float*)d_in[10];

    float* outMain = (float*)d_out;                                  // [B][S][H]
    float* attnF   = (float*)d_out + (size_t)NB * SS * HH;           // [NH*B][S][S]

    const int M = NB * SS;  // 8192
    char* w = (char*)d_ws;
    auto alloc = [&](size_t bytes) { char* r = w; w += (bytes + 255) & ~(size_t)255; return r; };
    __bf16* memBf   = (__bf16*)alloc((size_t)M * HH * 2);
    __bf16* concatA = (__bf16*)alloc((size_t)M * 2 * HH * 2);
    __bf16* WkT = (__bf16*)alloc((size_t)HH * HH * 2);
    __bf16* WvT = (__bf16*)alloc((size_t)HH * HH * 2);
    __bf16* WqT = (__bf16*)alloc((size_t)HH * HH * 2);
    __bf16* WfT = (__bf16*)alloc((size_t)2 * HH * HH * 2);
    __bf16* Qh  = (__bf16*)alloc((size_t)16 * SS * DD * 2);
    __bf16* Kt  = (__bf16*)alloc((size_t)16 * DD * SS * 2);
    __bf16* Vh  = (__bf16*)alloc((size_t)16 * SS * DD * 2);
    __bf16* Pb  = (__bf16*)alloc((size_t)16 * SS * SS * 2);

    // 1) conversions / weight transposes
    int nact4 = M * HH / 4;
    cvt_bf16_kernel<<<nact4 / 256, 256, 0, stream>>>(memory, memBf, nact4);
    cvt_dec_kernel<<<nact4 / 256, 256, 0, stream>>>(decoder, concatA, nact4);
    transpose_w_kernel<<<(HH * HH) / 256, 256, 0, stream>>>(Wk, WkT, HH, HH);
    transpose_w_kernel<<<(HH * HH) / 256, 256, 0, stream>>>(Wv, WvT, HH, HH);
    transpose_w_kernel<<<(HH * HH) / 256, 256, 0, stream>>>(Wq, WqT, HH, HH);
    transpose_w_kernel<<<(2 * HH * HH) / 256, 256, 0, stream>>>(Wf, WfT, HH, 2 * HH);

    // 2) Q/K/V projections (K written transposed per head)
    {
        GemmP g{}; g.K = HH;
        dim3 grid(HH / BN, M / BM, 1);
        g.A = concatA; g.lda = 2 * HH; g.Bm = WqT; g.ldb = HH; g.outBf = Qh;
        gemm_bf16_kernel<EPI_HEAD><<<grid, 256, 0, stream>>>(g);
        g.A = memBf; g.lda = HH; g.Bm = WvT; g.outBf = Vh;
        gemm_bf16_kernel<EPI_HEAD><<<grid, 256, 0, stream>>>(g);
        g.Bm = WkT; g.outBf = Kt;
        gemm_bf16_kernel<EPI_HEAD_T><<<grid, 256, 0, stream>>>(g);
    }

    // 3) scores = Q·K^T / 16, masked -> attns region of d_out
    {
        GemmP g{}; g.K = DD;
        g.A = Qh;  g.lda = DD; g.sAz = (size_t)SS * DD;
        g.Bm = Kt; g.ldb = SS; g.sBz = (size_t)DD * SS;
        g.outF = attnF; g.mask = mask;
        dim3 grid(SS / BN, SS / BM, 16);
        gemm_bf16_kernel<EPI_SCORES><<<grid, 256, 0, stream>>>(g);
    }

    // 4) softmax + query_mask; bf16 probs to ws
    softmax_kernel<<<16 * SS, 256, 0, stream>>>(attnF, Pb, qmask);

    // 5) ctx = P·V -> second half of concat buffer
    {
        GemmP g{}; g.K = SS;
        g.A = Pb;  g.lda = SS; g.sAz = (size_t)SS * SS;
        g.Bm = Vh; g.ldb = DD; g.sBz = (size_t)SS * DD;
        g.outBf = concatA;
        dim3 grid(DD / BN, SS / BM, 16);
        gemm_bf16_kernel<EPI_CTX><<<grid, 256, 0, stream>>>(g);
    }

    // 6) out = concat·Wf^T + bf + residual, then LayerNorm in place
    {
        GemmP g{}; g.K = 2 * HH;
        g.A = concatA; g.lda = 2 * HH; g.Bm = WfT; g.ldb = HH;
        g.outF = outMain; g.bias = bf; g.resid = decoder;
        dim3 grid(HH / BN, M / BM, 1);
        gemm_bf16_kernel<EPI_FINAL><<<grid, 256, 0, stream>>>(g);
    }
    layernorm_kernel<<<M, 256, 0, stream>>>(outMain, lns, lnb);
}
